// DNC_44959717654529
// MI455X (gfx1250) — matrix operand order repaired and checked by reading, not compiled, not run on hardware
//
#include <hip/hip_runtime.h>
#include <hip/hip_bf16.h>

// ---------------------------------------------------------------------------
// DNC forward for MI455X (gfx1250, wave32).
// GEMMs use v_wmma_f32_16x16x32_f16 (f16 in, f32 accum). Weights are packed
// once per launch into WMMA B-fragment tile layout. The large gates GEMM
// stages B tiles into LDS via GLOBAL_LOAD_ASYNC_TO_LDS_B128 (double-buffered,
// ASYNCcnt-tracked) and reuses each A fragment across 4 N-tiles. Attention/
// softmax is VALU + LDS with wave32 shuffle reductions. The memory tensor
// lives in its d_out slot and is updated in place each step.
// ---------------------------------------------------------------------------

#define BQ   64
#define LQ   256
#define EQ   512
#define HQ   1024
#define MQ   2048
#define HSQ  64
#define RQ   4
#define OQ   1024
#define CIQ  768
#define POQ  1280
#define G4   4096            // 4*H
#define KG   1792            // CI + H
#define KT_G 56              // KG/32
#define NT_G 256             // 4096/16
#define GNW  4               // N-tiles per block in gates GEMM
#define NHEAD 512            // padded head output cols
#define KT_H 32              // 1024/32
#define NT_H 32              // 512/16
#define KT_P 40              // 1280/32
#define NT_P 64              // 1024/16
#define EPSQ 1e-5f

typedef __attribute__((ext_vector_type(16))) _Float16 v16h;
typedef __attribute__((ext_vector_type(8)))  float    v8f;

// ---------------- fragment packing helpers (16-bit WMMA layouts) -----------
// A 16x32 f16 (ISA 7.12.2): lanes 0-15 hold K={0..7,16..23}, lanes 16-31 hold
// K={8..15,24..31} for row M = lane&15.
__device__ __forceinline__ void put_A(_Float16* buf, int KT, int m, int k, float v) {
  int mt = m >> 4, kt = k >> 5, kk = k & 31;
  int s = (kk >> 3) & 1;
  int e = (kk & 7) + (((kk >> 4) & 1) << 3);
  int lane = (m & 15) + (s << 4);
  buf[((size_t)(mt * KT + kt) * 32 + lane) * 16 + e] = (_Float16)v;
}
// B 32x16 f16: lanes 0-15 hold K={0..7,16..23}, lanes 16-31 hold
// K={8..15,24..31} for column N = lane&15 (same K order as A).
__device__ __forceinline__ void put_B(_Float16* buf, int KT, int n, int k, float v) {
  int nt = n >> 4, kt = k >> 5, kk = k & 31;
  int lane = (n & 15) + (((kk >> 3) & 1) << 4);
  int e = (kk & 7) + (((kk >> 4) & 1) << 3);
  buf[((size_t)(nt * KT + kt) * 32 + lane) * 16 + e] = (_Float16)v;
}

// ---------------- wave32 / block reductions --------------------------------
__device__ __forceinline__ float wave_sum(float v) {
  for (int off = 16; off; off >>= 1) v += __shfl_xor(v, off, 32);
  return v;
}
__device__ __forceinline__ float wave_max(float v) {
  for (int off = 16; off; off >>= 1) v = fmaxf(v, __shfl_xor(v, off, 32));
  return v;
}
__device__ __forceinline__ float block_sum(float v, float* s) {
  int tid = threadIdx.x;
  v = wave_sum(v);
  if ((tid & 31) == 0) s[tid >> 5] = v;
  __syncthreads();
  int nw = (blockDim.x + 31) >> 5;
  if (tid < 32) {
    float x = (tid < nw) ? s[tid] : 0.f;
    x = wave_sum(x);
    if (tid == 0) s[0] = x;
  }
  __syncthreads();
  float r = s[0];
  __syncthreads();
  return r;
}
__device__ __forceinline__ float sigf(float x) { return 1.f / (1.f + expf(-x)); }

// ---------------- weight packing (once per launch) -------------------------
__global__ void pack_gates_k(const float* __restrict__ w_ih, const float* __restrict__ w_hh,
                             const float* __restrict__ b_ih, const float* __restrict__ b_hh,
                             _Float16* __restrict__ W, float* __restrict__ bg) {
  long long idx = (long long)blockIdx.x * blockDim.x + threadIdx.x;
  if (idx >= (long long)G4 * KG) return;
  int n = (int)(idx / KG), k = (int)(idx % KG);
  float v = (k < CIQ) ? w_ih[(size_t)n * CIQ + k] : w_hh[(size_t)n * HQ + (k - CIQ)];
  put_B(W, KT_G, n, k, v);
  if (k == 0) bg[n] = b_ih[n] + b_hh[n];
}

__global__ void pack_heads_k(const float* __restrict__ w_rk, const float* __restrict__ b_rk,
                             const float* __restrict__ w_wk, const float* __restrict__ b_wk,
                             const float* __restrict__ w_ws, const float* __restrict__ b_ws,
                             const float* __restrict__ w_er, const float* __restrict__ b_er,
                             const float* __restrict__ w_ad, const float* __restrict__ b_ad,
                             _Float16* __restrict__ W, float* __restrict__ bh) {
  int idx = blockIdx.x * blockDim.x + threadIdx.x;
  if (idx >= NHEAD * HQ) return;
  int n = idx >> 10, k = idx & 1023;
  float v = 0.f, bias = 0.f;
  if (n < 256)       { v = w_rk[(size_t)n * HQ + k];        bias = b_rk[n]; }
  else if (n < 320)  { v = w_wk[(size_t)(n-256) * HQ + k];  bias = b_wk[n-256]; }
  else if (n < 384)  { v = w_er[(size_t)(n-320) * HQ + k];  bias = b_er[n-320]; }
  else if (n < 448)  { v = w_ad[(size_t)(n-384) * HQ + k];  bias = b_ad[n-384]; }
  else if (n == 448) { v = w_ws[k];                          bias = b_ws[0]; }
  put_B(W, KT_H, n, k, v);
  if (k == 0) bh[n] = bias;
}

__global__ void pack_proj_k(const float* __restrict__ w_p, _Float16* __restrict__ W) {
  long long idx = (long long)blockIdx.x * blockDim.x + threadIdx.x;
  if (idx >= (long long)OQ * POQ) return;
  int n = (int)(idx / POQ), k = (int)(idx % POQ);
  put_B(W, KT_P, n, k, w_p[idx]);
}

__global__ void zero_f32_k(float* p, long long n) {
  long long i = (long long)blockIdx.x * blockDim.x + threadIdx.x;
  if (i < n) p[i] = 0.f;
}
__global__ void zero_f16_k(_Float16* p, long long n) {
  long long i = (long long)blockIdx.x * blockDim.x + threadIdx.x;
  if (i < n) p[i] = (_Float16)0.f;
}

// ---------------- gates GEMM with async LDS staging ------------------------
// M=64, N=4096, K=1792. Block = 128 threads = 4 waves (wave = M-tile).
// Block covers GNW=4 consecutive N-tiles; wave mt DMA-stages B tile (nt0+mt)
// for each K-step into a double-buffered LDS slab via
// global_load_async_to_lds_b128 (ASYNCcnt), then all waves consume all 4
// tiles from LDS (ds_load_b128), reusing one A fragment per 4 WMMAs.
__global__ void __launch_bounds__(128)
gemm_gates_lds_k(const _Float16* __restrict__ A, const _Float16* __restrict__ Bw,
                 const float* __restrict__ bias, float* __restrict__ C) {
  __shared__ _Float16 Bsh[2][GNW][512];   // 2 x 4 x 1KB
  int nt0 = blockIdx.x * GNW;
  int mt = threadIdx.x >> 5;
  int lane = threadIdx.x & 31;
  const v16h* Af = (const v16h*)A + (size_t)mt * KT_G * 32 + lane;
  // global source of this wave's staged tile stream (tile nt0+mt, kt=0..)
  const char* gsrc = (const char*)(Bw + (size_t)(nt0 + mt) * KT_G * 512) + lane * 16;

  // stage B tile for K-step kt into buffer `buf` (each lane copies 2 x 16B)
  auto stage = [&](int buf, int kt) {
    unsigned l0 = (unsigned)(uintptr_t)(&Bsh[buf][mt][0]) + (unsigned)lane * 16u;
    const char* g0 = gsrc + (size_t)kt * 1024;
    asm volatile("global_load_async_to_lds_b128 %0, %1, off"
                 :: "v"(l0), "v"(g0) : "memory");
    asm volatile("global_load_async_to_lds_b128 %0, %1, off"
                 :: "v"(l0 + 512u), "v"(g0 + 512) : "memory");
  };

  stage(0, 0);
  asm volatile("s_wait_asynccnt 0x0" ::: "memory");
  __syncthreads();

  v8f acc[GNW] = {};
  for (int kt = 0; kt < KT_G; ++kt) {
    int cur = kt & 1, nxt = cur ^ 1;
    if (kt + 1 < KT_G) stage(nxt, kt + 1);      // prefetch next K-step
    v16h a = Af[kt * 32];                        // A fragment (reused 4x)
    for (int j = 0; j < GNW; ++j) {
      v16h b = *(const v16h*)&Bsh[cur][j][lane * 16];
      acc[j] = __builtin_amdgcn_wmma_f32_16x16x32_f16(false, a, false, b,
                                                      (short)0, acc[j], false, false);
    }
    if (kt + 1 < KT_G) asm volatile("s_wait_asynccnt 0x0" ::: "memory");
    __syncthreads();
  }

  int rb = (lane >> 4) * 8;
  for (int j = 0; j < GNW; ++j) {
    int ncol = (nt0 + j) * 16 + (lane & 15);
    float bs = bias[ncol];
    for (int r = 0; r < 8; ++r) {
      int m = mt * 16 + rb + r;
      C[(size_t)m * G4 + ncol] = acc[j][r] + bs;
    }
  }
}

// ---------------- generic direct WMMA GEMM (heads / projection) ------------
__global__ void gemm_wmma_k(const _Float16* __restrict__ A, const _Float16* __restrict__ Bw,
                            const float* __restrict__ bias, float* __restrict__ C,
                            int KT, long long row_stride) {
  int nt = blockIdx.x;
  int mt = threadIdx.x >> 5;
  int lane = threadIdx.x & 31;
  const v16h* Af = (const v16h*)A  + (size_t)mt * KT * 32 + lane;
  const v16h* Bf = (const v16h*)Bw + (size_t)nt * KT * 32 + lane;
  v8f acc = {};
  for (int kt = 0; kt < KT; ++kt) {
    __builtin_prefetch(Bf + (kt + 8) * 32, 0, 1);   // global_prefetch_b8
    v16h a = Af[kt * 32];
    v16h b = Bf[kt * 32];
    acc = __builtin_amdgcn_wmma_f32_16x16x32_f16(false, a, false, b,
                                                 (short)0, acc, false, false);
  }
  int ncol = nt * 16 + (lane & 15);
  float bs = bias ? bias[ncol] : 0.f;
  int rb = (lane >> 4) * 8;
  for (int r = 0; r < 8; ++r) {
    int m = mt * 16 + rb + r;
    C[(size_t)m * row_stride + ncol] = acc[r] + bs;
  }
}

// ---------------- per-step kernels -----------------------------------------
// ci = LN(concat(x_t, rv)) -> A fragments for K=[0,768)
__global__ void prep_A_k(const float* __restrict__ x_emb, int t, const float* __restrict__ rv,
                         const float* __restrict__ g, const float* __restrict__ bb,
                         _Float16* __restrict__ Ag) {
  __shared__ float s[8];
  int b = blockIdx.x, tid = threadIdx.x;
  float v[3];
  for (int i = 0; i < 3; ++i) {
    int j = tid + i * 256;
    v[i] = (j < EQ) ? x_emb[((size_t)b * LQ + t) * EQ + j]
                    : rv[(size_t)b * (RQ * HSQ) + (j - EQ)];
  }
  float mu = block_sum(v[0] + v[1] + v[2], s) * (1.f / CIQ);
  float ss = 0.f;
  for (int i = 0; i < 3; ++i) { float d = v[i] - mu; ss += d * d; }
  float rs = rsqrtf(block_sum(ss, s) * (1.f / CIQ) + EPSQ);
  for (int i = 0; i < 3; ++i) {
    int j = tid + i * 256;
    put_A(Ag, KT_G, b, j, (v[i] - mu) * rs * g[j] + bb[j]);
  }
}

// LSTM elementwise; also packs raw h into A fragments for K=[768,1792)
__global__ void lstm_k(const float* __restrict__ gates, float* __restrict__ c,
                       float* __restrict__ h, _Float16* __restrict__ Ag) {
  int idx = blockIdx.x * blockDim.x + threadIdx.x;
  if (idx >= BQ * HQ) return;
  int b = idx >> 10, j = idx & 1023;
  const float* gr = gates + (size_t)b * G4;
  float gi = gr[j], gf = gr[HQ + j], gg = gr[2 * HQ + j], go = gr[3 * HQ + j];
  float cn = sigf(gf) * c[idx] + sigf(gi) * tanhf(gg);
  float hv = sigf(go) * tanhf(cn);
  c[idx] = cn;
  h[idx] = hv;
  put_A(Ag, KT_G, b, CIQ + j, hv);
}

// hn = LN(h) -> f32 buffer + A fragments for head GEMM
__global__ void ln_hn_k(const float* __restrict__ h, const float* __restrict__ g,
                        const float* __restrict__ bb, float* __restrict__ hn,
                        _Float16* __restrict__ Ah) {
  __shared__ float s[8];
  int b = blockIdx.x, tid = threadIdx.x;
  float v[4];
  for (int i = 0; i < 4; ++i) v[i] = h[(size_t)b * HQ + tid + i * 256];
  float mu = block_sum(v[0] + v[1] + v[2] + v[3], s) * (1.f / HQ);
  float ss = 0.f;
  for (int i = 0; i < 4; ++i) { float d = v[i] - mu; ss += d * d; }
  float rs = rsqrtf(block_sum(ss, s) * (1.f / HQ) + EPSQ);
  for (int i = 0; i < 4; ++i) {
    int j = tid + i * 256;
    float y = (v[i] - mu) * rs * g[j] + bb[j];
    hn[(size_t)b * HQ + j] = y;
    put_A(Ah, KT_H, b, j, y);
  }
}

// 64-thread LN over one 64-vector
__device__ __forceinline__ void ln64(float x, int h, int lane, int wid, float* s2,
                                     const float* gg, const float* bbv, float* outp) {
  float sm = wave_sum(x);
  if (lane == 0) s2[wid] = sm;
  __syncthreads();
  float mu = (s2[0] + s2[1]) * (1.f / 64.f);
  __syncthreads();
  float d = x - mu;
  float vs = wave_sum(d * d);
  if (lane == 0) s2[wid] = vs;
  __syncthreads();
  float var = (s2[0] + s2[1]) * (1.f / 64.f);
  __syncthreads();
  outp[h] = d * rsqrtf(var + EPSQ) * gg[h] + bbv[h];
}

// head activations + key LayerNorms
__global__ void heads_post_k(const float* __restrict__ ho,
                             const float* __restrict__ wkg, const float* __restrict__ wkb,
                             const float* __restrict__ rkg, const float* __restrict__ rkb,
                             float* __restrict__ wk_n, float* __restrict__ rk_n,
                             float* __restrict__ er, float* __restrict__ ad,
                             float* __restrict__ wstr) {
  __shared__ float s2[2];
  int b = blockIdx.x, h = threadIdx.x, lane = h & 31, wid = h >> 5;
  const float* row = ho + (size_t)b * NHEAD;
  er[(size_t)b * HSQ + h] = sigf(row[320 + h]);
  ad[(size_t)b * HSQ + h] = tanhf(row[384 + h]);
  if (h == 0) wstr[b] = sigf(row[448]);
  ln64(row[256 + h], h, lane, wid, s2, wkg, wkb, wk_n + (size_t)b * HSQ);
  for (int r = 0; r < RQ; ++r)
    ln64(row[r * 64 + h], h, lane, wid, s2, rkg, rkb, rk_n + (size_t)(b * RQ + r) * HSQ);
}

// write attention: scores = LN(mem).wk_n, softmax*wstr, in-place erase/add
__global__ void write_attn_k(float* __restrict__ mem, const float* __restrict__ wk_n,
                             const float* __restrict__ er, const float* __restrict__ ad,
                             const float* __restrict__ wstr,
                             const float* __restrict__ mg, const float* __restrict__ mb) {
  __shared__ float sc[MQ];
  __shared__ float red[8];
  int b = blockIdx.x, tid = threadIdx.x, lane = tid & 31, wid = tid >> 5;
  float g0 = mg[lane], g1 = mg[lane + 32], bb0 = mb[lane], bb1 = mb[lane + 32];
  float wk0 = wk_n[(size_t)b * HSQ + lane], wk1 = wk_n[(size_t)b * HSQ + lane + 32];
  float* memb = mem + (size_t)b * MQ * HSQ;
  float lmax = -3.4e38f;
  for (int m = wid; m < MQ; m += 8) {
    float v0 = memb[(size_t)m * HSQ + lane];
    float v1 = memb[(size_t)m * HSQ + lane + 32];
    float mu = wave_sum(v0 + v1) * (1.f / 64.f);
    float d0 = v0 - mu, d1 = v1 - mu;
    float rs = rsqrtf(wave_sum(d0 * d0 + d1 * d1) * (1.f / 64.f) + EPSQ);
    float dot = wave_sum((d0 * rs * g0 + bb0) * wk0 + (d1 * rs * g1 + bb1) * wk1);
    if (lane == 0) sc[m] = dot;
    lmax = fmaxf(lmax, dot);
  }
  lmax = wave_max(lmax);
  if (lane == 0) red[wid] = lmax;
  __syncthreads();
  float gmax = red[0];
  for (int i = 1; i < 8; ++i) gmax = fmaxf(gmax, red[i]);
  __syncthreads();
  float lsum = 0.f;
  for (int m = tid; m < MQ; m += 256) {
    float e = expf(sc[m] - gmax);
    sc[m] = e;
    lsum += e;
  }
  lsum = wave_sum(lsum);
  __syncthreads();
  if (lane == 0) red[wid] = lsum;
  __syncthreads();
  float gsum = 0.f;
  for (int i = 0; i < 8; ++i) gsum += red[i];
  float scale = wstr[b] / gsum;
  float e0 = er[(size_t)b * HSQ + lane], e1 = er[(size_t)b * HSQ + lane + 32];
  float a0 = ad[(size_t)b * HSQ + lane], a1 = ad[(size_t)b * HSQ + lane + 32];
  for (int m = wid; m < MQ; m += 8) {
    float w = sc[m] * scale;
    size_t i0 = (size_t)m * HSQ + lane;
    float v0 = memb[i0], v1 = memb[i0 + 32];
    memb[i0]      = v0 * (1.f - w * e0) + w * a0;
    memb[i0 + 32] = v1 * (1.f - w * e1) + w * a1;
  }
}

// read attention: 4 heads, softmax over m, rv = rw @ mem
__global__ void read_attn_k(const float* __restrict__ mem, const float* __restrict__ rk_n,
                            const float* __restrict__ mg, const float* __restrict__ mb,
                            float* __restrict__ rv) {
  __shared__ float sc[RQ][MQ];     // 32 KB
  __shared__ float racc[RQ][HSQ];
  __shared__ float red[8];
  int b = blockIdx.x, tid = threadIdx.x, lane = tid & 31, wid = tid >> 5;
  const float* memb = mem + (size_t)b * MQ * HSQ;
  float g0 = mg[lane], g1 = mg[lane + 32], bb0 = mb[lane], bb1 = mb[lane + 32];
  float rk0[RQ], rk1[RQ], lmax[RQ];
  for (int r = 0; r < RQ; ++r) {
    rk0[r] = rk_n[(size_t)(b * RQ + r) * HSQ + lane];
    rk1[r] = rk_n[(size_t)(b * RQ + r) * HSQ + lane + 32];
    lmax[r] = -3.4e38f;
  }
  for (int m = wid; m < MQ; m += 8) {
    float v0 = memb[(size_t)m * HSQ + lane];
    float v1 = memb[(size_t)m * HSQ + lane + 32];
    float mu = wave_sum(v0 + v1) * (1.f / 64.f);
    float d0 = v0 - mu, d1 = v1 - mu;
    float rs = rsqrtf(wave_sum(d0 * d0 + d1 * d1) * (1.f / 64.f) + EPSQ);
    float n0 = d0 * rs * g0 + bb0, n1 = d1 * rs * g1 + bb1;
    for (int r = 0; r < RQ; ++r) {
      float dot = wave_sum(n0 * rk0[r] + n1 * rk1[r]);
      if (lane == 0) sc[r][m] = dot;
      lmax[r] = fmaxf(lmax[r], dot);
    }
  }
  for (int i = tid; i < RQ * HSQ; i += 256) ((float*)racc)[i] = 0.f;
  float inv[RQ];
  for (int r = 0; r < RQ; ++r) {
    float wm = wave_max(lmax[r]);
    if (lane == 0) red[wid] = wm;
    __syncthreads();
    float gmax = red[0];
    for (int i = 1; i < 8; ++i) gmax = fmaxf(gmax, red[i]);
    __syncthreads();
    float ls = 0.f;
    for (int m = tid; m < MQ; m += 256) {
      float e = expf(sc[r][m] - gmax);
      sc[r][m] = e;
      ls += e;
    }
    ls = wave_sum(ls);
    __syncthreads();
    if (lane == 0) red[wid] = ls;
    __syncthreads();
    float gs = 0.f;
    for (int i = 0; i < 8; ++i) gs += red[i];
    inv[r] = 1.f / gs;
    __syncthreads();
  }
  float acc0[RQ] = {}, acc1[RQ] = {};
  for (int m = wid; m < MQ; m += 8) {
    size_t i0 = (size_t)m * HSQ + lane;
    float v0 = memb[i0], v1 = memb[i0 + 32];
    for (int r = 0; r < RQ; ++r) {
      float w = sc[r][m] * inv[r];
      acc0[r] += w * v0;
      acc1[r] += w * v1;
    }
  }
  for (int r = 0; r < RQ; ++r) {
    atomicAdd(&racc[r][lane], acc0[r]);
    atomicAdd(&racc[r][lane + 32], acc1[r]);
  }
  __syncthreads();
  for (int i = tid; i < RQ * HSQ; i += 256)
    rv[(size_t)b * (RQ * HSQ) + i] = ((float*)racc)[i];
}

// out = LN(concat(hn, rv)) -> A fragments for projection GEMM
__global__ void out_prep_k(const float* __restrict__ hn, const float* __restrict__ rv,
                           const float* __restrict__ g, const float* __restrict__ bb,
                           _Float16* __restrict__ Ap) {
  __shared__ float s[8];
  int b = blockIdx.x, tid = threadIdx.x;
  float v[5];
  for (int i = 0; i < 5; ++i) {
    int j = tid + i * 256;
    v[i] = (j < HQ) ? hn[(size_t)b * HQ + j] : rv[(size_t)b * (RQ * HSQ) + (j - HQ)];
  }
  float mu = block_sum(v[0] + v[1] + v[2] + v[3] + v[4], s) * (1.f / POQ);
  float ss = 0.f;
  for (int i = 0; i < 5; ++i) { float d = v[i] - mu; ss += d * d; }
  float rs = rsqrtf(block_sum(ss, s) * (1.f / POQ) + EPSQ);
  for (int i = 0; i < 5; ++i) {
    int j = tid + i * 256;
    put_A(Ap, KT_P, b, j, (v[i] - mu) * rs * g[j] + bb[j]);
  }
}

__global__ void finalize_k(const float* __restrict__ h, const float* __restrict__ c,
                           float* __restrict__ hT, float* __restrict__ cT) {
  int i = blockIdx.x * blockDim.x + threadIdx.x;
  if (i < BQ * HQ) { hT[i] = h[i]; cT[i] = c[i]; }
}

// ---------------------------------------------------------------------------
extern "C" void kernel_launch(void* const* d_in, const int* in_sizes, int n_in,
                              void* d_out, int out_size, void* d_ws, size_t ws_size,
                              hipStream_t stream) {
  const float* x_emb  = (const float*)d_in[0];
  const float* ln_in_g= (const float*)d_in[1];
  const float* ln_in_b= (const float*)d_in[2];
  const float* w_ih   = (const float*)d_in[3];
  const float* w_hh   = (const float*)d_in[4];
  const float* b_ih   = (const float*)d_in[5];
  const float* b_hh   = (const float*)d_in[6];
  const float* ln_c_g = (const float*)d_in[7];
  const float* ln_c_b = (const float*)d_in[8];
  const float* w_rk   = (const float*)d_in[9];
  const float* b_rk   = (const float*)d_in[10];
  const float* w_wk   = (const float*)d_in[11];
  const float* b_wk   = (const float*)d_in[12];
  const float* w_ws   = (const float*)d_in[13];
  const float* b_ws   = (const float*)d_in[14];
  const float* w_er   = (const float*)d_in[15];
  const float* b_er   = (const float*)d_in[16];
  const float* w_ad   = (const float*)d_in[17];
  const float* b_ad   = (const float*)d_in[18];
  const float* ln_rk_g= (const float*)d_in[19];
  const float* ln_rk_b= (const float*)d_in[20];
  const float* ln_wk_g= (const float*)d_in[21];
  const float* ln_wk_b= (const float*)d_in[22];
  const float* ln_m_g = (const float*)d_in[23];
  const float* ln_m_b = (const float*)d_in[24];
  const float* ln_o_g = (const float*)d_in[25];
  const float* ln_o_b = (const float*)d_in[26];
  const float* w_p    = (const float*)d_in[27];
  const float* b_p    = (const float*)d_in[28];

  // output layout: outputs[B,L,O] | memT[B,M,HS] | hT[B,H] | cT[B,H]
  float* out_logits = (float*)d_out;
  float* mem = out_logits + (size_t)BQ * LQ * OQ;
  float* hT  = mem + (size_t)BQ * MQ * HSQ;
  float* cT  = hT + (size_t)BQ * HQ;

  // workspace carve-up (256B aligned)
  size_t off = 0;
  auto take = [&](size_t bytes) -> char* {
    char* p = (char*)d_ws + off;
    off = (off + bytes + 255) & ~(size_t)255;
    return p;
  };
  _Float16* WgP = (_Float16*)take((size_t)G4 * KG * 2);
  _Float16* WhP = (_Float16*)take((size_t)NHEAD * HQ * 2);
  _Float16* WpP = (_Float16*)take((size_t)OQ * POQ * 2);
  _Float16* Ag  = (_Float16*)take((size_t)4 * KT_G * 512 * 2);
  _Float16* Ah  = (_Float16*)take((size_t)4 * KT_H * 512 * 2);
  _Float16* Ap  = (_Float16*)take((size_t)4 * KT_P * 512 * 2);
  float* bgate   = (float*)take(G4 * 4);
  float* bhead   = (float*)take(NHEAD * 4);
  float* gates   = (float*)take((size_t)BQ * G4 * 4);
  float* h       = (float*)take((size_t)BQ * HQ * 4);
  float* c       = (float*)take((size_t)BQ * HQ * 4);
  float* hn      = (float*)take((size_t)BQ * HQ * 4);
  float* rv      = (float*)take((size_t)BQ * RQ * HSQ * 4);
  float* headout = (float*)take((size_t)BQ * NHEAD * 4);
  float* wk_n    = (float*)take((size_t)BQ * HSQ * 4);
  float* rk_n    = (float*)take((size_t)BQ * RQ * HSQ * 4);
  float* er      = (float*)take((size_t)BQ * HSQ * 4);
  float* ad      = (float*)take((size_t)BQ * HSQ * 4);
  float* wstr    = (float*)take((size_t)BQ * 4);

  // --- weight packing + state init -----------------------------------------
  {
    long long n = (long long)G4 * KG;
    pack_gates_k<<<(unsigned)((n + 255) / 256), 256, 0, stream>>>(w_ih, w_hh, b_ih, b_hh, WgP, bgate);
  }
  pack_heads_k<<<(NHEAD * HQ + 255) / 256, 256, 0, stream>>>(
      w_rk, b_rk, w_wk, b_wk, w_ws, b_ws, w_er, b_er, w_ad, b_ad, WhP, bhead);
  {
    long long n = (long long)OQ * POQ;
    pack_proj_k<<<(unsigned)((n + 255) / 256), 256, 0, stream>>>(w_p, WpP);
  }
  zero_f32_k<<<(BQ * HQ + 255) / 256, 256, 0, stream>>>(h, BQ * HQ);
  zero_f32_k<<<(BQ * HQ + 255) / 256, 256, 0, stream>>>(c, BQ * HQ);
  zero_f32_k<<<(BQ * RQ * HSQ + 255) / 256, 256, 0, stream>>>(rv, BQ * RQ * HSQ);
  {
    long long n = (long long)BQ * MQ * HSQ;
    zero_f32_k<<<(unsigned)((n + 255) / 256), 256, 0, stream>>>(mem, n);
  }
  {
    long long n = (long long)4 * KT_G * 512;  // h-region of Ag must start at h0=0
    zero_f16_k<<<(unsigned)((n + 255) / 256), 256, 0, stream>>>(Ag, n);
  }

  // --- sequential scan over timesteps --------------------------------------
  for (int t = 0; t < LQ; ++t) {
    prep_A_k<<<BQ, 256, 0, stream>>>(x_emb, t, rv, ln_in_g, ln_in_b, Ag);
    gemm_gates_lds_k<<<NT_G / GNW, 128, 0, stream>>>(Ag, WgP, bgate, gates);
    lstm_k<<<(BQ * HQ + 255) / 256, 256, 0, stream>>>(gates, c, h, Ag);
    ln_hn_k<<<BQ, 256, 0, stream>>>(h, ln_c_g, ln_c_b, hn, Ah);
    gemm_wmma_k<<<NT_H, 128, 0, stream>>>(Ah, WhP, bhead, headout, KT_H, (long long)NHEAD);
    heads_post_k<<<BQ, 64, 0, stream>>>(headout, ln_wk_g, ln_wk_b, ln_rk_g, ln_rk_b,
                                        wk_n, rk_n, er, ad, wstr);
    write_attn_k<<<BQ, 256, 0, stream>>>(mem, wk_n, er, ad, wstr, ln_m_g, ln_m_b);
    read_attn_k<<<BQ, 256, 0, stream>>>(mem, rk_n, ln_m_g, ln_m_b, rv);
    out_prep_k<<<BQ, 256, 0, stream>>>(hn, rv, ln_o_g, ln_o_b, Ap);
    gemm_wmma_k<<<NT_P, 128, 0, stream>>>(Ap, WpP, b_p, out_logits + (size_t)t * OQ,
                                          KT_P, (long long)LQ * OQ);
  }
  finalize_k<<<(BQ * HQ + 255) / 256, 256, 0, stream>>>(h, c, hT, cT);
}
